// ProjectionBlock_71528385347749
// MI455X (gfx1250) — matrix operand order repaired and checked by reading, not compiled, not run on hardware
//
#include <hip/hip_runtime.h>

// ---------------------------------------------------------------------------
// ProjectionBlock: Y[b] = W @ X[b]  (100x400 @ 400x10), then QR(Y[b]) -> Q
// B=8192, D_IN=400, D_OUT=100, Q=10
// GEMM recast as one big fp16 WMMA GEMM W(100x400) @ X'(400x81920) with fp32
// accumulate; N axis = flattened (batch, q). fp16 > bf16 here: data is unit
// scale (X~N(0,1), W~N(0,1)/20) so fp16's 3 extra mantissa bits are free
// accuracy. QR done per-batch in fp32 (MGS) per wave.
// Software-pipelined X staging (double-buffered LDS, one barrier per K-block),
// all staging address math hoisted; NT hints on streaming loads/stores.
// ---------------------------------------------------------------------------

typedef __attribute__((ext_vector_type(16))) _Float16 v16h;
typedef __attribute__((ext_vector_type(8)))  float    v8f;
typedef __attribute__((ext_vector_type(4)))  float    v4f;

#define BATCH    8192
#define D_IN     400
#define D_OUT    100
#define QDIM     10
#define NB       8                 // batches per workgroup
#define NCOL     (NB * QDIM)       // 80 columns of Y' per WG = 5 N-tiles
#define KB       13                // ceil(400/32) K-blocks
#define WB_M     128               // padded W rows (8 M-tiles, waves 0..7)
#define WB_K     416               // padded W cols (13 K-blocks)
#define XS_STRIDE 40               // halfs per staged column (32 used, 16B-aligned)
#define XS_SIZE  (NCOL * XS_STRIDE)
#define YS_STRIDE 101              // odd stride -> conflict-free column access
#define E4_PER_KB 640              // float4s per staged K-block (2560 f32)

// --- prologue: W (100x400 f32) -> Wh (128x416 f16, zero padded) in d_ws -----
__global__ void frmap_wconv(const float* __restrict__ W,
                            _Float16* __restrict__ Wh) {
  int idx = blockIdx.x * blockDim.x + threadIdx.x;
  if (idx >= WB_M * WB_K) return;
  int row = idx / WB_K, k = idx % WB_K;
  float v = (row < D_OUT && k < D_IN) ? W[row * D_IN + k] : 0.0f;
  Wh[idx] = (_Float16)v;
}

// --- main: fused GEMM (fp16 WMMA, fp32 acc) + per-batch MGS QR --------------
__global__ __launch_bounds__(256) void frmap_qr(
    const float* __restrict__ X,           // (B, 400, 10) f32
    const _Float16* __restrict__ Wh,       // (128, 416) f16, zero padded
    float* __restrict__ out)               // (B, 100, 10) f32
{
  __shared__ __align__(16) _Float16 Xs[2][XS_SIZE]; // 2 x 6.4 KB
  __shared__ float Ys[NCOL * YS_STRIDE];            // 32.3 KB

  const int tid   = threadIdx.x;
  const int wave  = tid >> 5;      // 0..7 -> M-tile (and QR batch) for this wave
  const int lane  = tid & 31;
  const int lhalf = lane >> 4;     // K-half select per ISA A/B layouts
  const int l16   = lane & 15;
  const long long batch0 = (long long)blockIdx.x * NB;
  const float* Xbase = X + batch0 * (D_IN * QDIM);

  v8f acc[5] = { {}, {}, {}, {}, {} };       // 5 N-tiles x (16x16 f32 C)
  const int mrow = wave * 16 + l16;          // A-matrix row for this lane
  // A fragment addressing (16-bit A layout): lanes 0-15 K {0..7,16..23},
  // lanes 16-31 K {8..15,24..31}; f16 row stride 416 -> 16B aligned chunks.
  const uint4* ap_base = (const uint4*)(Wh + (size_t)mrow * WB_K + lhalf * 8);
  // B fragment LDS base (half index), K-block invariant
  const int bq_base = l16 * XS_STRIDE + lhalf * 8;

  // ---- staging state, ALL K-block invariant (hoisted out of the loop) ----
  // slots 0,1 always active (512 float4s); slot 2 active for tid < 128
  const float* gp[3];    // global base per float4 slot (advance = +320 f32/blk)
  int  ldso[12];         // LDS half offsets for the 12 staged elements
  int  iend[3];          // last i_local touched by the float4
  const bool sact2 = (tid + 512) < E4_PER_KB;
#pragma unroll
  for (int it = 0; it < 3; ++it) {
    int e4   = tid + it * 256;
    int g    = e4 / 80;
    int rem  = (e4 % 80) * 4;
    gp[it]   = Xbase + g * (D_IN * QDIM) + rem;
    iend[it] = (rem + 3) / QDIM;
#pragma unroll
    for (int j = 0; j < 4; ++j) {
      int e = rem + j;
      ldso[it * 4 + j] = (g * QDIM + e % QDIM) * XS_STRIDE + e / QDIM;
    }
  }

  float R[12];                               // in-flight staged f32 values

  // ---- pipeline prologue: issue global loads for K-block 0 (NT: read-once) --
#pragma unroll
  for (int it = 0; it < 3; ++it) {
    v4f t = {0.f, 0.f, 0.f, 0.f};
    if (it < 2 || sact2) t = __builtin_nontemporal_load((const v4f*)gp[it]);
    R[it * 4 + 0] = t.x; R[it * 4 + 1] = t.y;
    R[it * 4 + 2] = t.z; R[it * 4 + 3] = t.w;
  }

#pragma unroll
  for (int kb = 0; kb < KB; ++kb) {
    _Float16* buf = &Xs[kb & 1][0];

    // ---- convert & store staged block kb into LDS (f16, column-major) ----
#pragma unroll
    for (int it = 0; it < 3; ++it) {
      if (it < 2 || sact2) {
#pragma unroll
        for (int j = 0; j < 4; ++j)
          buf[ldso[it * 4 + j]] = (_Float16)R[it * 4 + j];
      }
    }
    __syncthreads();   // block kb visible to all; block kb-2 reads all retired

    // ---- issue global loads for block kb+1 (overlaps WMMA below) ----
    if (kb + 1 < KB) {
#pragma unroll
      for (int it = 0; it < 3; ++it) {
        // full blocks for kb+1 < 12; at kb+1 == 12 only i_local < 16 is valid
        bool ok  = (it < 2 || sact2) && ((kb + 1) < (KB - 1) || iend[it] < 16);
        v4f t = {0.f, 0.f, 0.f, 0.f};
        if (ok) t = __builtin_nontemporal_load(
                        (const v4f*)(gp[it] + (kb + 1) * 32 * QDIM));
        R[it * 4 + 0] = t.x; R[it * 4 + 1] = t.y;
        R[it * 4 + 2] = t.z; R[it * 4 + 3] = t.w;
      }
      // warm L2 two blocks ahead (global_prefetch_b8)
      if (kb + 2 < KB)
        __builtin_prefetch(gp[0] + (kb + 2) * 32 * QDIM, 0, 1);
    }

    // ---- A fragment for this K-block: 2 x b128 from L2-resident Wh ----
    union { uint4 u[2]; v16h v; } A;
    A.u[0] = ap_base[kb * 4 + 0];   // K base + {0..7}
    A.u[1] = ap_base[kb * 4 + 2];   // K base + {16..23}

#pragma unroll
    for (int s = 0; s < 5; ++s) {
      // B fragment: lane = column c; lanes 0-15 K {0..7,16..23}, lanes 16-31 K {8..15,24..31}
      union { uint4 u[2]; v16h v; } Bf;
      const uint4* bp = (const uint4*)&buf[bq_base + s * 16 * XS_STRIDE];
      Bf.u[0] = bp[0];
      Bf.u[1] = bp[2];
      acc[s] = __builtin_amdgcn_wmma_f32_16x16x32_f16(
          false, A.v, false, Bf.v, (short)0, acc[s], false, false);
    }
  }

  // ---- scatter accumulators into Ys (column-major, odd stride) ----
  // C layout: lane n<16 -> N=n, VGPR r -> M = 16*wave + r; lanes 16-31 -> M+8
#pragma unroll
  for (int s = 0; s < 5; ++s) {
    int c     = s * 16 + l16;
    int rbase = wave * 16 + lhalf * 8;
#pragma unroll
    for (int r = 0; r < 8; ++r) {
      int row = rbase + r;
      if (row < D_OUT) Ys[c * YS_STRIDE + row] = acc[s][r];
    }
  }
  __syncthreads();

  // ---- per-batch Modified Gram-Schmidt QR (fp32, register resident) ----
  {
    const int cbase = wave * QDIM;           // wave w handles batch0 + w
    float v[QDIM][4];
#pragma unroll
    for (int j = 0; j < QDIM; ++j)
#pragma unroll
      for (int s2 = 0; s2 < 4; ++s2) {
        int row = lane + 32 * s2;
        v[j][s2] = (row < D_OUT) ? Ys[(cbase + j) * YS_STRIDE + row] : 0.0f;
      }

#pragma unroll
    for (int j = 0; j < QDIM; ++j) {
      float d = 0.0f;
#pragma unroll
      for (int s2 = 0; s2 < 4; ++s2) d += v[j][s2] * v[j][s2];
#pragma unroll
      for (int off = 16; off > 0; off >>= 1) d += __shfl_xor(d, off, 32);
      float inv = rsqrtf(d);
#pragma unroll
      for (int s2 = 0; s2 < 4; ++s2) v[j][s2] *= inv;
      for (int k = j + 1; k < QDIM; ++k) {
        float r = 0.0f;
#pragma unroll
        for (int s2 = 0; s2 < 4; ++s2) r += v[j][s2] * v[k][s2];
#pragma unroll
        for (int off = 16; off > 0; off >>= 1) r += __shfl_xor(r, off, 32);
#pragma unroll
        for (int s2 = 0; s2 < 4; ++s2) v[k][s2] -= r * v[j][s2];
      }
    }

    // write orthonormal columns back through LDS, then stream out coalesced
#pragma unroll
    for (int j = 0; j < QDIM; ++j)
#pragma unroll
      for (int s2 = 0; s2 < 4; ++s2) {
        int row = lane + 32 * s2;
        if (row < D_OUT) Ys[(cbase + j) * YS_STRIDE + row] = v[j][s2];
      }
    // dense, coalesced, write-once (NT) output: e = row*10 + j incremental
    float* ob = out + (batch0 + wave) * (D_OUT * QDIM);
    int e = lane, row = lane / QDIM, j = lane % QDIM;
    while (e < D_OUT * QDIM) {
      __builtin_nontemporal_store(Ys[(cbase + j) * YS_STRIDE + row], &ob[e]);
      e += 32; row += 3; j += 2;
      if (j >= QDIM) { j -= QDIM; row += 1; }
    }
  }
}

extern "C" void kernel_launch(void* const* d_in, const int* in_sizes, int n_in,
                              void* d_out, int out_size, void* d_ws, size_t ws_size,
                              hipStream_t stream) {
  (void)in_sizes; (void)n_in; (void)out_size; (void)ws_size;
  const float* X = (const float*)d_in[0];   // (8192, 400, 10) f32
  const float* W = (const float*)d_in[1];   // (100, 400) f32
  float* out = (float*)d_out;               // (8192, 100, 10) f32
  _Float16* Wh = (_Float16*)d_ws;           // 128*416 f16 = 104 KB scratch

  frmap_wconv<<<(WB_M * WB_K + 255) / 256, 256, 0, stream>>>(W, Wh);
  frmap_qr<<<BATCH / NB, 256, 0, stream>>>(X, Wh, out);
}
